// Decoder_Attention_48387101556746
// MI455X (gfx1250) — matrix operand order repaired and checked by reading, not compiled, not run on hardware
//
#include <hip/hip_runtime.h>

// Problem dims (fixed by the reference).
#define B_ 128
#define T_ 32
#define S_ 256
#define H_ 1024
#define E_ 1024

typedef __attribute__((ext_vector_type(16))) __bf16 v16bf;
typedef __attribute__((ext_vector_type(8)))  __bf16 v8bf;
typedef __attribute__((ext_vector_type(8)))  float  v8f;
typedef __attribute__((ext_vector_type(4)))  unsigned int u32x4;
typedef __attribute__((ext_vector_type(8)))  int i32x8;
typedef __attribute__((ext_vector_type(4)))  int i32x4;

#define HAVE_TDM __has_builtin(__builtin_amdgcn_tensor_load_to_lds)

// ---------- bf16 helpers (bit-exact RNE) ----------
__device__ __forceinline__ unsigned short f2bf_bits(float f) {
    union { float f; unsigned u; } x; x.f = f;
    unsigned r = x.u + 0x7FFFu + ((x.u >> 16) & 1u);
    return (unsigned short)(r >> 16);
}
__device__ __forceinline__ __bf16 f2bf(float f) {
    unsigned short hs = f2bf_bits(f);
    return __builtin_bit_cast(__bf16, hs);
}
__device__ __forceinline__ float bf2f(__bf16 h) {
    unsigned short hs = __builtin_bit_cast(unsigned short, h);
    union { unsigned u; float f; } x; x.u = ((unsigned)hs) << 16;
    return x.f;
}

// ---------- WMMA wrapper ----------
__device__ __forceinline__ v8f wmma_bf16_f32(v16bf a, v16bf b, v8f c) {
    return __builtin_amdgcn_wmma_f32_16x16x32_bf16(false, a, false, b, (short)0, c, false, false);
}

// A fragment from LDS tile [16][40] (80B rows, 16B aligned).
__device__ __forceinline__ v16bf frag_a_lds(const __bf16* As, int lane) {
    const int lda = 40;
    int row = lane & 15;
    int kb  = (lane < 16) ? 0 : 8;
    const __bf16* p = As + row * lda + kb;
    v8bf lo = *(const v8bf*)(p);        // ds_load_b128
    v8bf hi = *(const v8bf*)(p + 16);   // ds_load_b128
    v16bf a;
#pragma unroll
    for (int i = 0; i < 8; ++i) { a[i] = lo[i]; a[8 + i] = hi[i]; }
    return a;
}

// B fragment from transposed LDS tile BsT[16][40]: BsT[n][k] = B[k][n].
__device__ __forceinline__ v16bf frag_b_lds(const __bf16* BsT, int lane) {
    const int ldb = 40;
    int col = lane & 15;
    int kb  = (lane < 16) ? 0 : 8;
    const __bf16* p = BsT + col * ldb + kb;
    v8bf lo = *(const v8bf*)(p);        // ds_load_b128
    v8bf hi = *(const v8bf*)(p + 16);   // ds_load_b128
    v16bf b;
#pragma unroll
    for (int i = 0; i < 8; ++i) { b[i] = lo[i]; b[8 + i] = hi[i]; }
    return b;
}

#if HAVE_TDM
// TDM: load a [128 rows x 32 cols] bf16 tile from WT (row stride K elements) into
// LDS at ldsOff, padding each 64B row by 16B (-> ld = 40 elements).
__device__ __forceinline__ void tdm_load_b_tile(const __bf16* gsrc, unsigned ldsOff,
                                                unsigned K, unsigned Nrows) {
    unsigned long long ga = (unsigned long long)(size_t)gsrc;
    u32x4 g0;
    g0[0] = 1u;                                     // count=1 (valid), user mode
    g0[1] = ldsOff;                                 // lds_addr (bytes)
    g0[2] = (unsigned)ga;                           // global_addr[31:0]
    g0[3] = (unsigned)((ga >> 32) & 0x01FFFFFFull)  // global_addr[56:32]
          | (2u << 30);                             // type=2 (image)
    i32x8 g1;
    g1[0] = (int)((1u << 16)    // data_size = 1 (2 bytes)
                | (1u << 20)    // pad_enable
                | (3u << 22)    // pad_interval: 16 DWORDs = 64B
                | (3u << 25));  // pad_amount: 4 DWORDs = 16B
    g1[1] = (int)((K & 0xFFFFu) << 16);                               // tensor_dim0 lo
    g1[2] = (int)(((K >> 16) & 0xFFFFu) | ((Nrows & 0xFFFFu) << 16)); // dim0 hi | dim1 lo
    g1[3] = (int)(((Nrows >> 16) & 0xFFFFu) | (32u << 16));           // dim1 hi | tile_dim0=32
    g1[4] = (int)128;                                                 // tile_dim1=128
    g1[5] = (int)K;                                                   // tensor_dim0_stride lo
    g1[6] = 0;
    g1[7] = 0;
    i32x4 z4 = {0, 0, 0, 0};
#if __clang_major__ >= 23
    i32x8 z8 = {0, 0, 0, 0, 0, 0, 0, 0};
    __builtin_amdgcn_tensor_load_to_lds(g0, g1, z4, z4, z8, 0);
#else
    __builtin_amdgcn_tensor_load_to_lds(g0, g1, z4, z4, 0);
#endif
}
#endif

// ---------- double-buffered LDS tiles ----------
struct alignas(16) SmemBuf {
    __bf16 As[2][64][40];     // 2 x 5120 B
    __bf16 BsT[2][128][40];   // 2 x 10240 B
};

// Stage a 64x32 fp32->bf16 A tile (fully unrolled: 4 x global_load_b64 per thread).
template <typename AF>
__device__ __forceinline__ void stage_a(AF aload, __bf16 (*As)[40], int kk, int tid) {
#pragma unroll
    for (int u = 0; u < 4; ++u) {
        int i = u * 256 + tid;
        int r = i >> 4, c2 = (i & 15) * 2;
        float f0 = aload(r, kk + c2);
        float f1 = aload(r, kk + c2 + 1);
        unsigned pk = ((unsigned)f2bf_bits(f1) << 16) | f2bf_bits(f0);
        *(unsigned*)&As[r][c2] = pk;
    }
}

#if !HAVE_TDM
__device__ __forceinline__ void stage_b_manual(__bf16 (*BsT)[40], const __bf16* WT,
                                               int nBase, int kk, int K, int tid) {
#pragma unroll
    for (int u = 0; u < 16; ++u) {
        int i = u * 256 + tid;
        int n = i >> 5, k = i & 31;
        BsT[n][k] = WT[(size_t)(nBase + n) * K + kk + k];
    }
}
#endif

// Double-buffered pipelined GEMM core: block tile 64(M) x 128(N), wave = 16x64.
// Per K-step: issue TDM for step s+1, run 4 WMMAs on step s (DMA in flight),
// stage next A tile, wait tensorcnt, single barrier.
template <typename AF>
__device__ __forceinline__ void gemm_pipeline(SmemBuf* sm, const __bf16* WT, int K,
                                              int Ntot, int nBase, AF aload,
                                              int tid, int lane, int wid,
                                              int wm, int wn, v8f* acc) {
    const int nsteps = K / 32;
    // prologue: fill buffer 0
#if HAVE_TDM
    if (wid == 0)
        tdm_load_b_tile(WT + (size_t)nBase * K, (unsigned)(size_t)&sm->BsT[0][0][0], K, Ntot);
#else
    stage_b_manual(sm->BsT[0], WT, nBase, 0, K, tid);
#endif
    stage_a(aload, sm->As[0], 0, tid);
#if HAVE_TDM
    if (wid == 0) __builtin_amdgcn_s_wait_tensorcnt(0);
#endif
    __syncthreads();

    for (int s = 0; s < nsteps; ++s) {
        const int cur = s & 1, nxt = cur ^ 1;
        const int kkn = (s + 1) * 32;
        const bool hasNext = (s + 1 < nsteps);
#if HAVE_TDM
        if (hasNext && wid == 0)
            tdm_load_b_tile(WT + (size_t)nBase * K + kkn,
                            (unsigned)(size_t)&sm->BsT[nxt][0][0], K, Ntot);
#endif
        // compute current step (overlaps with TDM of next step)
        v16bf af = frag_a_lds(&sm->As[cur][wm * 16][0], lane);
#pragma unroll
        for (int j = 0; j < 4; ++j) {
            v16bf bf = frag_b_lds(&sm->BsT[cur][wn * 64 + j * 16][0], lane);
            acc[j] = wmma_bf16_f32(af, bf, acc[j]);
        }
        if (hasNext) {
#if !HAVE_TDM
            stage_b_manual(sm->BsT[nxt], WT, nBase, kkn, K, tid);
#endif
            stage_a(aload, sm->As[nxt], kkn, tid);
        }
#if HAVE_TDM
        if (hasNext && wid == 0) __builtin_amdgcn_s_wait_tensorcnt(0);
#endif
        __syncthreads();
    }
}

// ---------- weight convert + transpose: src fp32 [K][N] -> dst bf16 [N][K] ----------
__global__ __launch_bounds__(256) void xpose_bf16(const float* __restrict__ src,
                                                  __bf16* __restrict__ dst, int K, int N) {
    __shared__ float tile[32][33];
    int kb = blockIdx.y * 32, nb = blockIdx.x * 32;
    int tx = threadIdx.x & 31, ty = threadIdx.x >> 5;  // 32x8
    for (int i = ty; i < 32; i += 8)
        tile[i][tx] = src[(size_t)(kb + i) * N + nb + tx];
    __syncthreads();
    for (int i = ty; i < 32; i += 8)
        dst[(size_t)(nb + i) * K + kb + tx] = f2bf(tile[tx][i]);
}

// =======================================================================
// GEMM kernels
// =======================================================================

// encP = enc @ W2  (M = B*S, N = H, K = H), bf16 out
__global__ __launch_bounds__(256) void encp_gemm(const float* __restrict__ enc,
                                                 const __bf16* __restrict__ W2T,
                                                 __bf16* __restrict__ encP) {
    __shared__ SmemBuf sm;
    const int tid = threadIdx.x, lane = tid & 31, wid = tid >> 5;
    const int wm = wid >> 1, wn = wid & 1;
    const int mBase = blockIdx.y * 64;
    const int nBase = blockIdx.x * 128;
    v8f acc[4] = {{}, {}, {}, {}};
    auto aload = [=](int r, int k) -> float {
        return enc[(size_t)(mBase + r) * H_ + k];
    };
    gemm_pipeline(&sm, W2T, H_, H_, nBase, aload, tid, lane, wid, wm, wn, acc);

    const int rb = (lane < 16) ? 0 : 8;
    const int row0 = mBase + wm * 16 + rb;
#pragma unroll
    for (int j = 0; j < 4; ++j) {
        int col = nBase + wn * 64 + j * 16 + (lane & 15);
#pragma unroll
        for (int r = 0; r < 8; ++r)
            encP[(size_t)(row0 + r) * H_ + col] = f2bf(acc[j][r]);
    }
}

// hP = h @ W1  (M = B, N = H, K = H), fp32 out
__global__ __launch_bounds__(256) void hw_gemm(const float* __restrict__ h,
                                               const __bf16* __restrict__ W1T,
                                               float* __restrict__ hP) {
    __shared__ SmemBuf sm;
    const int tid = threadIdx.x, lane = tid & 31, wid = tid >> 5;
    const int wm = wid >> 1, wn = wid & 1;
    const int mBase = blockIdx.y * 64;
    const int nBase = blockIdx.x * 128;
    v8f acc[4] = {{}, {}, {}, {}};
    auto aload = [=](int r, int k) -> float {
        return h[(size_t)(mBase + r) * H_ + k];
    };
    gemm_pipeline(&sm, W1T, H_, H_, nBase, aload, tid, lane, wid, wm, wn, acc);

    const int rb = (lane < 16) ? 0 : 8;
    const int row0 = mBase + wm * 16 + rb;
#pragma unroll
    for (int j = 0; j < 4; ++j) {
        int col = nBase + wn * 64 + j * 16 + (lane & 15);
#pragma unroll
        for (int r = 0; r < 8; ++r)
            hP[(size_t)(row0 + r) * H_ + col] = acc[j][r];
    }
}

// gi = [x_t, ctx] @ Wi ; rc = h @ Uh  (M = B, N = 3H)
__global__ __launch_bounds__(256) void gru_gemm(const float* __restrict__ dec,
                                                const float* __restrict__ ctx,
                                                const float* __restrict__ h,
                                                const __bf16* __restrict__ WiT,
                                                const __bf16* __restrict__ UhT,
                                                float* __restrict__ gi,
                                                float* __restrict__ rc, int t) {
    __shared__ SmemBuf sm;
    const int tid = threadIdx.x, lane = tid & 31, wid = tid >> 5;
    const int wm = wid >> 1, wn = wid & 1;
    const int mBase = blockIdx.y * 64;       // batch rows
    const int nBase = blockIdx.x * 128;      // columns of 3H
    const int N = 3 * H_;

    v8f acc1[4] = {{}, {}, {}, {}};
    auto aload1 = [=](int r, int k) -> float {
        return (k < E_) ? dec[((size_t)(mBase + r) * T_ + t) * E_ + k]
                        : ctx[(size_t)(mBase + r) * H_ + (k - E_)];
    };
    gemm_pipeline(&sm, WiT, E_ + H_, N, nBase, aload1, tid, lane, wid, wm, wn, acc1);

    v8f acc2[4] = {{}, {}, {}, {}};
    auto aload2 = [=](int r, int k) -> float {
        return h[(size_t)(mBase + r) * H_ + k];
    };
    gemm_pipeline(&sm, UhT, H_, N, nBase, aload2, tid, lane, wid, wm, wn, acc2);

    const int rb = (lane < 16) ? 0 : 8;
    const int row0 = mBase + wm * 16 + rb;
#pragma unroll
    for (int j = 0; j < 4; ++j) {
        int col = nBase + wn * 64 + j * 16 + (lane & 15);
#pragma unroll
        for (int r = 0; r < 8; ++r) {
            gi[(size_t)(row0 + r) * N + col] = acc1[j][r];
            rc[(size_t)(row0 + r) * N + col] = acc2[j][r];
        }
    }
}

// ---------- fused score/softmax/context per batch row ----------
__global__ __launch_bounds__(256) void attn_kernel(const float* __restrict__ hP,
                                                   const __bf16* __restrict__ encP,
                                                   const float* __restrict__ V,
                                                   const float* __restrict__ enc,
                                                   float* __restrict__ ctx) {
    const int b = blockIdx.x;
    const int tid = threadIdx.x, lane = tid & 31, wid = tid >> 5;
    __shared__ float sc[256];
    __shared__ float red[256];
    const float*  hPb   = hP + (size_t)b * H_;
    const __bf16* encPb = encP + (size_t)b * S_ * H_;

    // scores: each wave handles 32 source positions; lanes own 8-wide vector chunks
    for (int j = 0; j < 32; ++j) {
        int s = wid * 32 + j;
        const __bf16* row = encPb + (size_t)s * H_;
        float acc = 0.f;
#pragma unroll
        for (int c = 0; c < 4; ++c) {
            int h0 = lane * 8 + c * 256;
            v8bf e8 = *(const v8bf*)(row + h0);   // global_load_b128
#pragma unroll
            for (int i = 0; i < 8; ++i)
                acc += tanhf(hPb[h0 + i] + bf2f(e8[i])) * V[h0 + i];
        }
#pragma unroll
        for (int off = 16; off > 0; off >>= 1)
            acc += __shfl_xor(acc, off);
        if (lane == 0) sc[s] = acc;
    }
    __syncthreads();

    // softmax over S=256
    red[tid] = sc[tid];
    __syncthreads();
    for (int off = 128; off > 0; off >>= 1) {
        if (tid < off) red[tid] = fmaxf(red[tid], red[tid + off]);
        __syncthreads();
    }
    float m = red[0];
    __syncthreads();
    float e = __expf(sc[tid] - m);
    red[tid] = e;
    __syncthreads();
    for (int off = 128; off > 0; off >>= 1) {
        if (tid < off) red[tid] += red[tid + off];
        __syncthreads();
    }
    float inv = 1.f / red[0];
    __syncthreads();
    sc[tid] = e * inv;
    __syncthreads();

    // context[b,h] = sum_s attn[s] * enc[b,s,h]  (coalesced across threads)
    const float* encb = enc + (size_t)b * S_ * H_;
#pragma unroll
    for (int c = 0; c < 4; ++c) {
        int h = c * 256 + tid;
        float acc = 0.f;
        for (int s = 0; s < S_; ++s)
            acc += sc[s] * encb[(size_t)s * H_ + h];
        ctx[(size_t)b * H_ + h] = acc;
    }
}

// ---------- GRU gate fusion + state update + output write ----------
__global__ __launch_bounds__(256) void gru_gate(const float* __restrict__ gi,
                                                const float* __restrict__ rc,
                                                const float* __restrict__ bias,
                                                float* __restrict__ h,
                                                float* __restrict__ out, int t) {
    int idx = blockIdx.x * 256 + threadIdx.x;   // over B*H
    int b = idx >> 10;
    int n = idx & (H_ - 1);
    const float* gib = gi + (size_t)b * 3 * H_;
    const float* rcb = rc + (size_t)b * 3 * H_;
    float gz = gib[n]      + bias[n]      + rcb[n];
    float gr = gib[H_ + n] + bias[H_ + n] + rcb[H_ + n];
    float z = 1.f / (1.f + __expf(-gz));
    float r = 1.f / (1.f + __expf(-gr));
    float hh = tanhf(gib[2 * H_ + n] + bias[2 * H_ + n] + r * rcb[2 * H_ + n]);
    float hold = h[idx];
    float hnew = z * hold + (1.f - z) * hh;
    h[idx] = hnew;
    out[((size_t)b * T_ + t) * H_ + n] = hnew;
}

extern "C" void kernel_launch(void* const* d_in, const int* in_sizes, int n_in,
                              void* d_out, int out_size, void* d_ws, size_t ws_size,
                              hipStream_t stream) {
    const float* dec  = (const float*)d_in[0];   // [B,T,E]
    const float* enc  = (const float*)d_in[1];   // [B,S,H]
    const float* h0   = (const float*)d_in[2];   // [B,H]
    const float* Watt = (const float*)d_in[3];   // [2H,H]
    const float* Vatt = (const float*)d_in[4];   // [H,1]
    const float* Wi   = (const float*)d_in[5];   // [E+H,3H]
    const float* Uh   = (const float*)d_in[6];   // [H,3H]
    const float* bias = (const float*)d_in[7];   // [3H]
    float* out = (float*)d_out;

    char* ws = (char*)d_ws;
    size_t off = 0;
    auto wsAlloc = [&](size_t bytes) -> void* {
        void* p = ws + off;
        off += (bytes + 255) & ~(size_t)255;
        return p;
    };
    __bf16* encP = (__bf16*)wsAlloc((size_t)B_ * S_ * H_ * 2);       // 64 MiB, L2-resident
    __bf16* W1T  = (__bf16*)wsAlloc((size_t)H_ * H_ * 2);            // [N][K] transposed
    __bf16* W2T  = (__bf16*)wsAlloc((size_t)H_ * H_ * 2);
    __bf16* WiT  = (__bf16*)wsAlloc((size_t)3 * H_ * (E_ + H_) * 2);
    __bf16* UhT  = (__bf16*)wsAlloc((size_t)3 * H_ * H_ * 2);
    float*  h_cur = (float*)wsAlloc((size_t)B_ * H_ * 4);
    float*  hP    = (float*)wsAlloc((size_t)B_ * H_ * 4);
    float*  ctx   = (float*)wsAlloc((size_t)B_ * H_ * 4);
    float*  gi    = (float*)wsAlloc((size_t)B_ * 3 * H_ * 4);
    float*  rc    = (float*)wsAlloc((size_t)B_ * 3 * H_ * 4);

    hipMemcpyAsync(h_cur, h0, (size_t)B_ * H_ * 4, hipMemcpyDeviceToDevice, stream);

    // Convert + transpose weights to bf16 [N][K] so B tiles are TDM-friendly.
    xpose_bf16<<<dim3(H_ / 32, H_ / 32), 256, 0, stream>>>(Watt,            W1T, H_, H_);
    xpose_bf16<<<dim3(H_ / 32, H_ / 32), 256, 0, stream>>>(Watt + H_ * H_,  W2T, H_, H_);
    xpose_bf16<<<dim3(3 * H_ / 32, (E_ + H_) / 32), 256, 0, stream>>>(Wi, WiT, E_ + H_, 3 * H_);
    xpose_bf16<<<dim3(3 * H_ / 32, H_ / 32), 256, 0, stream>>>(Uh, UhT, H_, 3 * H_);

    // Time-invariant half of the Bahdanau score: encP = enc @ W_att[H:2H]
    encp_gemm<<<dim3(H_ / 128, (B_ * S_) / 64), 256, 0, stream>>>(enc, W2T, encP);

    for (int t = 0; t < T_; ++t) {
        hw_gemm<<<dim3(H_ / 128, B_ / 64), 256, 0, stream>>>(h_cur, W1T, hP);
        attn_kernel<<<B_, 256, 0, stream>>>(hP, encP, Vatt, enc, ctx);
        gru_gemm<<<dim3(3 * H_ / 128, B_ / 64), 256, 0, stream>>>(dec, ctx, h_cur, WiT, UhT, gi, rc, t);
        gru_gate<<<(B_ * H_) / 256, 256, 0, stream>>>(gi, rc, bias, h_cur, out, t);
    }
}
